// ContentBasedAttention_79190607004261
// MI455X (gfx1250) — matrix operand order repaired and checked by reading, not compiled, not run on hardware
//
#include <hip/hip_runtime.h>
#include <hip/hip_bf16.h>
#include <stdint.h>

// Problem constants (fixed by the reference)
#define HD 512
#define SEQ 2048
#define NB 64
#define NCHUNK 4
#define CHUNK_S (SEQ / NCHUNK)   // 512 rows of S per workgroup
#define SBLK 64                  // rows per s-block tile
#define KEYS_STRIDE 520          // halfs, padded (512+8) -> conflict-free A frags
#define PANEL_STRIDE 40          // halfs, padded (32+8)  -> conflict-free B frags
#define PART_STRIDE 514          // per (b,chunk) partial: m, l, ctx[512]

typedef __attribute__((ext_vector_type(16))) __bf16 bf16x16;
typedef __attribute__((ext_vector_type(8)))  float  f32x8;

union FragU { bf16x16 v; uint4 q[2]; };

__device__ __forceinline__ unsigned short f2bf(float f) {
  unsigned int u = __builtin_bit_cast(unsigned int, f);
  u = (u + 0x7FFFu + ((u >> 16) & 1u)) >> 16;  // round-to-nearest-even
  return (unsigned short)u;
}
__device__ __forceinline__ float bf2f(unsigned short h) {
  unsigned int u = ((unsigned int)h) << 16;
  return __builtin_bit_cast(float, u);
}
__device__ __forceinline__ float fast_tanh(float x) {
  // tanh(x) = 1 - 2/(exp(2x)+1); __expf saturates so limits are correct +-1
  return 1.0f - 2.0f / (__expf(2.0f * x) + 1.0f);
}

// ---------------- prep: W_s fp32 -> bf16 (layout unchanged: [o][h], h contiguous == B[k][n] K-contig) ----
__global__ void ws_to_bf16_kernel(const float* __restrict__ W_s,
                                  unsigned short* __restrict__ wsb) {
  int i = (blockIdx.x * 256 + threadIdx.x) * 4;
  float4 f = *(const float4*)(W_s + i);
  ushort4 h = make_ushort4(f2bf(f.x), f2bf(f.y), f2bf(f.z), f2bf(f.w));
  *(ushort4*)(wsb + i) = h;
}

// ---------------- prep: qb[b][o] = b_s[o] + sum_h dec[b,h] * W_h[o,h] ----------------
__global__ void proj_q_kernel(const float* __restrict__ dec,
                              const float* __restrict__ W_h,
                              const float* __restrict__ b_s,
                              float* __restrict__ qb) {
  __shared__ float s_q[HD];
  int b = blockIdx.x, t = threadIdx.x;
  s_q[t] = dec[b * HD + t];
  s_q[t + 256] = dec[b * HD + t + 256];
  __syncthreads();
  for (int o = t; o < HD; o += 256) {
    const float* wr = W_h + (size_t)o * HD;
    float acc = b_s[o];
    #pragma unroll 4
    for (int h = 0; h < HD; h += 4) {
      float4 w = *(const float4*)(wr + h);
      acc += w.x * s_q[h] + w.y * s_q[h + 1] + w.z * s_q[h + 2] + w.w * s_q[h + 3];
    }
    qb[b * HD + o] = acc;
  }
}

// ---------------- main: fused proj_k GEMM (WMMA bf16) + energy + online softmax + context ----------------
__global__ void __launch_bounds__(256)
attn_main_kernel(const float* __restrict__ enc,
                 const unsigned short* __restrict__ wsb,
                 const float* __restrict__ qb,
                 const float* __restrict__ vvec,
                 float* __restrict__ part) {
  __shared__ unsigned short s_keys[SBLK * KEYS_STRIDE];  // 66560 B, keys block as bf16
  __shared__ unsigned short s_panel[HD * PANEL_STRIDE];  // 40960 B, W_s K-panel [n][k]
  __shared__ float s_energy[SBLK];
  __shared__ float s_p[SBLK];
  __shared__ float s_ctx[HD];

  const int t    = threadIdx.x;
  const int lane = t & 31;
  const int wid  = t >> 5;        // 0..7
  const int mt   = wid & 3;       // M-tile (16 rows) within the 64-row block
  const int nh   = wid >> 2;      // N-half: columns [nh*256, nh*256+256)
  const int l16  = lane & 15;
  const bool hi  = lane >= 16;

  const int chunk = blockIdx.x;   // 0..3
  const int b     = blockIdx.y;   // 0..63

  float m_run = -1e30f, l_run = 0.0f;
  s_ctx[t] = 0.0f;
  s_ctx[t + 256] = 0.0f;

  const size_t enc_base = ((size_t)b * SEQ + (size_t)chunk * CHUNK_S) * HD;

  // single LDS base pointers; all tile addressing folds into immediate offsets
  const unsigned short* ap_base = s_keys + (mt * 16 + l16) * KEYS_STRIDE + (hi ? 16 : 0);
  const unsigned short* bp_base = s_panel + (nh * 256 + l16) * PANEL_STRIDE + (hi ? 16 : 0);

  for (int sb = 0; sb < CHUNK_S / SBLK; ++sb) {
    __syncthreads();  // previous iteration done reading s_keys / s_energy
    if (t < SBLK) s_energy[t] = 0.0f;

    // ---- stage keys block: 64 x 512 fp32 -> bf16 in LDS (read once from HBM) ----
    const float* kb = enc + enc_base + (size_t)sb * SBLK * HD;
    #pragma unroll 8
    for (int it = 0; it < 32; ++it) {
      int c   = t + it * 256;       // 8192 float4 chunks
      int row = c >> 7;
      int cf  = (c & 127) * 4;
      float4 f = *(const float4*)(kb + (size_t)row * HD + cf);
      ushort4 h = make_ushort4(f2bf(f.x), f2bf(f.y), f2bf(f.z), f2bf(f.w));
      *(ushort4*)&s_keys[row * KEYS_STRIDE + cf] = h;
    }
    // kick off prefetch of the NEXT keys block into the cache hierarchy
    if (sb + 1 < CHUNK_S / SBLK) {
      const float* kn = kb + (size_t)SBLK * HD;
      #pragma unroll
      for (int q = 0; q < 4; ++q)  // 256 thr * 4 * 128B = 128 KB = whole next block
        __builtin_prefetch(kn + (size_t)(t + q * 256) * 32, 0, 1);
    }
    __syncthreads();

    // ---- GEMM: proj = keys(64x512) @ W_s^T(512x512), K streamed in 32-wide panels ----
    f32x8 acc[16];
    #pragma unroll
    for (int j = 0; j < 16; ++j)
      #pragma unroll
      for (int i = 0; i < 8; ++i) acc[j][i] = 0.0f;

    for (int kk = 0; kk < 16; ++kk) {
      __syncthreads();  // previous panel fully consumed
      #pragma unroll
      for (int q = 0; q < 8; ++q) {
        int c  = t + q * 256;       // 2048 16-byte chunks: n = c/4, quarter = c%4
        int n  = c >> 2;
        int p4 = c & 3;
        *(uint4*)&s_panel[n * PANEL_STRIDE + p4 * 8] =
            *(const uint4*)&wsb[(size_t)n * HD + kk * 32 + p4 * 8];
      }
      __syncthreads();

      // A fragment (16x32 bf16): lane<16 holds K 0..15, lane>=16 holds K 16..31
      FragU A;
      {
        const unsigned short* ap = ap_base + kk * 32;
        A.q[0] = *(const uint4*)(ap);
        A.q[1] = *(const uint4*)(ap + 8);
      }
      // B fragments double-buffered: tile j+1 loads in flight while WMMA j executes
      FragU Bf[2];
      Bf[0].q[0] = *(const uint4*)(bp_base);
      Bf[0].q[1] = *(const uint4*)(bp_base + 8);
      #pragma unroll
      for (int j = 0; j < 16; ++j) {
        if (j + 1 < 16) {
          const unsigned short* bn = bp_base + (j + 1) * 16 * PANEL_STRIDE;
          Bf[(j + 1) & 1].q[0] = *(const uint4*)(bn);
          Bf[(j + 1) & 1].q[1] = *(const uint4*)(bn + 8);
        }
        acc[j] = __builtin_amdgcn_wmma_f32_16x16x32_bf16(
            false, A.v, false, Bf[j & 1].v, (short)0, acc[j], false, false);
      }
    }

    // ---- energy: e[row] = sum_col v[col] * tanh(acc + qb[b][col]) ----
    // qb/v loaded HERE (L2-hot, once per s-block) so they are not live across the GEMM
    float ep[8];
    #pragma unroll
    for (int i = 0; i < 8; ++i) ep[i] = 0.0f;
    #pragma unroll
    for (int j = 0; j < 16; ++j) {
      int col  = nh * 256 + j * 16 + l16;
      float qv = qb[b * HD + col];
      float vv = vvec[col];
      #pragma unroll
      for (int i = 0; i < 8; ++i)
        ep[i] += vv * fast_tanh(acc[j][i] + qv);
    }

    // reduce over columns = over the 16 lanes of each half-wave
    #pragma unroll
    for (int i = 0; i < 8; ++i) {
      float e = ep[i];
      e += __shfl_xor(e, 1, 32);
      e += __shfl_xor(e, 2, 32);
      e += __shfl_xor(e, 4, 32);
      e += __shfl_xor(e, 8, 32);
      ep[i] = e;
    }
    if (l16 == 0) {
      int rbase = mt * 16 + (hi ? 8 : 0);   // C layout: lane half selects rows 0-7 / 8-15
      #pragma unroll
      for (int i = 0; i < 8; ++i) atomicAdd(&s_energy[rbase + i], ep[i]);
    }
    __syncthreads();

    // ---- online softmax + running context (keys read only from LDS) ----
    float m_blk = -1e30f;
    #pragma unroll 8
    for (int i = 0; i < SBLK; ++i) m_blk = fmaxf(m_blk, s_energy[i]);
    float m_new = fmaxf(m_run, m_blk);
    float scale = __expf(m_run - m_new);
    if (t < SBLK) s_p[t] = __expf(s_energy[t] - m_new);
    __syncthreads();
    float l_blk = 0.0f;
    #pragma unroll 8
    for (int i = 0; i < SBLK; ++i) l_blk += s_p[i];
    l_run = l_run * scale + l_blk;
    m_run = m_new;

    float c0 = s_ctx[t] * scale;
    float c1 = s_ctx[t + 256] * scale;
    #pragma unroll 4
    for (int i = 0; i < SBLK; ++i) {
      float p = s_p[i];
      c0 += p * bf2f(s_keys[i * KEYS_STRIDE + t]);
      c1 += p * bf2f(s_keys[i * KEYS_STRIDE + t + 256]);
    }
    s_ctx[t] = c0;
    s_ctx[t + 256] = c1;
  }

  // ---- write this chunk's partial (m, l, ctx[512]) ----
  float* pp = part + (size_t)(b * NCHUNK + chunk) * PART_STRIDE;
  if (t == 0) { pp[0] = m_run; pp[1] = l_run; }
  pp[2 + t] = s_ctx[t];
  pp[2 + t + 256] = s_ctx[t + 256];
}

// ---------------- finalize: merge 4 chunk partials per batch ----------------
__global__ void finalize_kernel(const float* __restrict__ part, float* __restrict__ out) {
  int b = blockIdx.x, t = threadIdx.x;
  const float* p = part + (size_t)b * NCHUNK * PART_STRIDE;
  float M = -1e30f;
  #pragma unroll
  for (int c = 0; c < NCHUNK; ++c) M = fmaxf(M, p[c * PART_STRIDE]);
  float w[NCHUNK];
  float L = 0.0f;
  #pragma unroll
  for (int c = 0; c < NCHUNK; ++c) {
    w[c] = __expf(p[c * PART_STRIDE] - M);
    L += w[c] * p[c * PART_STRIDE + 1];
  }
  float inv = 1.0f / L;
  for (int h = t; h < HD; h += 256) {
    float acc = 0.0f;
    #pragma unroll
    for (int c = 0; c < NCHUNK; ++c) acc += w[c] * p[c * PART_STRIDE + 2 + h];
    out[(size_t)b * HD + h] = acc * inv;
  }
}

extern "C" void kernel_launch(void* const* d_in, const int* in_sizes, int n_in,
                              void* d_out, int out_size, void* d_ws, size_t ws_size,
                              hipStream_t stream) {
  const float* enc = (const float*)d_in[0];  // [B,S,H]
  const float* dec = (const float*)d_in[1];  // [1,B,H]
  const float* W_h = (const float*)d_in[2];  // [H,H]
  const float* W_s = (const float*)d_in[3];  // [H,H]
  const float* b_s = (const float*)d_in[4];  // [H]
  const float* v   = (const float*)d_in[5];  // [H]
  float* out = (float*)d_out;                // [B,1,H]

  // workspace layout: W_s bf16 (512 KB) | qb fp32 (128 KB) | partials (~514 KB)
  unsigned short* wsb = (unsigned short*)d_ws;
  float* qbuf = (float*)((char*)d_ws + (size_t)HD * HD * sizeof(unsigned short));
  float* part = qbuf + (size_t)NB * HD;

  ws_to_bf16_kernel<<<(HD * HD) / 1024, 256, 0, stream>>>(W_s, wsb);
  proj_q_kernel<<<NB, 256, 0, stream>>>(dec, W_h, b_s, qbuf);
  dim3 grid(NCHUNK, NB);
  attn_main_kernel<<<grid, 256, 0, stream>>>(enc, wsb, qbuf, v, part);
  finalize_kernel<<<NB, 256, 0, stream>>>(part, out);
}
